// SwinTemporalBlock_22204980920602
// MI455X (gfx1250) — compile-verified, matrix operand order checked by reading, not run on hardware
//
#include <hip/hip_runtime.h>
#include <hip/hip_bf16.h>
#include <math.h>

// ---------------------------------------------------------------------------
// Types for CDNA5 WMMA (wave32, 16x16x32 bf16 -> f32)
// ---------------------------------------------------------------------------
typedef __attribute__((ext_vector_type(16))) __bf16 v16bf;
typedef __attribute__((ext_vector_type(8)))  float  v8f;
typedef __attribute__((ext_vector_type(4)))  int    v4i;

union Frag { uint4 u[2]; v16bf v; };

#define LDS_STRIDE 40   // 32 halves of data + 8 pad halves (80B row, kills bank conflicts)

#define AS1 __attribute__((address_space(1)))
#define AS3 __attribute__((address_space(3)))

#if defined(__has_builtin)
# if __has_builtin(__builtin_amdgcn_global_load_async_to_lds_b128)
#  define HAVE_ASYNC_B128 1
# endif
#endif
#ifndef HAVE_ASYNC_B128
# define HAVE_ASYNC_B128 0
#endif

__device__ __forceinline__ void wait_async0()
{
#if HAVE_ASYNC_B128
# if defined(__has_builtin) && __has_builtin(__builtin_amdgcn_s_wait_asynccnt)
    __builtin_amdgcn_s_wait_asynccnt(0);
# else
    asm volatile("s_wait_asynccnt 0x0" ::: "memory");
# endif
#endif
}

// Stage one 64-byte row chunk (32 bf16) from global into LDS.
__device__ __forceinline__ void stage_row(const __bf16* __restrict__ src,
                                          __bf16* __restrict__ dst)
{
#if HAVE_ASYNC_B128
    // CDNA5 async DMA: memory -> LDS without VGPR round trip (ASYNCcnt tracked)
    __builtin_amdgcn_global_load_async_to_lds_b128((AS1 v4i*)(src +  0), (AS3 v4i*)(dst +  0), 0, 0);
    __builtin_amdgcn_global_load_async_to_lds_b128((AS1 v4i*)(src +  8), (AS3 v4i*)(dst +  8), 0, 0);
    __builtin_amdgcn_global_load_async_to_lds_b128((AS1 v4i*)(src + 16), (AS3 v4i*)(dst + 16), 0, 0);
    __builtin_amdgcn_global_load_async_to_lds_b128((AS1 v4i*)(src + 24), (AS3 v4i*)(dst + 24), 0, 0);
#else
    const uint4* s4 = (const uint4*)src;
    uint4* d4 = (uint4*)dst;
    uint4 t0 = s4[0], t1 = s4[1], t2 = s4[2], t3 = s4[3];
    d4[0] = t0; d4[1] = t1; d4[2] = t2; d4[3] = t3;
#endif
}

__device__ __forceinline__ void stage_zero(__bf16* __restrict__ dst)
{
    uint4 z; z.x = z.y = z.z = z.w = 0u;
    uint4* d4 = (uint4*)dst;
    d4[0] = z; d4[1] = z; d4[2] = z; d4[3] = z;
}

// ---------------------------------------------------------------------------
// Generic bf16 GEMM: C[M,N] = A[M,K](bf16,row-major) @ Bt[N,K](bf16,row-major,
// i.e. B pre-transposed)  + bias (f32), optional exact-GELU, optional residual
// (f32), output f32 or bf16.
// Block: 128 threads = 4 waves (2x2), wave tile 64x64 (4x4 WMMA 16x16 tiles),
// K-step 32, ping-pong LDS staging via async global->LDS B128.
// ---------------------------------------------------------------------------
__launch_bounds__(128)
__global__ void gemm_bf16_kernel(const __bf16* __restrict__ A,
                                 const __bf16* __restrict__ Bt,
                                 const float*  __restrict__ bias,
                                 const float*  __restrict__ res,
                                 float*        __restrict__ outF,
                                 __bf16*       __restrict__ outH,
                                 int M, int N, int K, int act)
{
    __shared__ __attribute__((aligned(16))) __bf16 As[2][128 * LDS_STRIDE];
    __shared__ __attribute__((aligned(16))) __bf16 Bs[2][128 * LDS_STRIDE];

    const int tid  = threadIdx.x;
    const int lane = tid & 31;
    const int wave = tid >> 5;
    const int wm   = wave >> 1;
    const int wn   = wave & 1;
    const int hi   = lane >> 4;
    const int lr   = lane & 15;
    const int m0   = blockIdx.y * 128;
    const int n0   = blockIdx.x * 128;

    v8f acc[4][4] = {};

    auto loadTile = [&](int k0, int buf) {
        // A tile: thread t stages row m0+t (guard tail of M)
        {
            const int row = m0 + tid;
            __bf16* dst = &As[buf][tid * LDS_STRIDE];
            if (row < M) stage_row(A + (size_t)row * K + k0, dst);
            else         stage_zero(dst);
        }
        // B tile: thread t stages row n0+t of Bt (N is always a multiple of 128)
        {
            const int row = n0 + tid;
            stage_row(Bt + (size_t)row * K + k0, &Bs[buf][tid * LDS_STRIDE]);
        }
    };

    const int nk = K >> 5;
    loadTile(0, 0);
    wait_async0();
    __syncthreads();

    for (int kt = 0; kt < nk; ++kt) {
        const int cur = kt & 1;
        if (kt + 1 < nk) loadTile((kt + 1) << 5, cur ^ 1);
        if (kt + 2 < nk && (m0 + tid) < M)
            __builtin_prefetch(A + (size_t)(m0 + tid) * K + ((kt + 2) << 5), 0, 1);

        Frag a[4], b[4];
#pragma unroll
        for (int t = 0; t < 4; ++t) {
            const __bf16* pa = &As[cur][(wm * 64 + t * 16 + lr) * LDS_STRIDE + hi * 8];
            a[t].u[0] = *(const uint4*)pa;
            a[t].u[1] = *(const uint4*)(pa + 16);
            const __bf16* pb = &Bs[cur][(wn * 64 + t * 16 + lr) * LDS_STRIDE + hi * 8];
            b[t].u[0] = *(const uint4*)pb;
            b[t].u[1] = *(const uint4*)(pb + 16);
        }
#pragma unroll
        for (int mt = 0; mt < 4; ++mt)
#pragma unroll
            for (int nt = 0; nt < 4; ++nt)
                acc[mt][nt] = __builtin_amdgcn_wmma_f32_16x16x32_bf16(
                    false, a[mt].v, false, b[nt].v,
                    (short)0, acc[mt][nt], false, false);

        wait_async0();        // next tile's async DMA must be in LDS before barrier
        __syncthreads();
    }

    // Epilogue. C/D layout: VGPR r, lane: M = r + 8*hi (in tile), N = lr.
#pragma unroll
    for (int mt = 0; mt < 4; ++mt) {
#pragma unroll
        for (int nt = 0; nt < 4; ++nt) {
            const int col = n0 + wn * 64 + nt * 16 + lr;
#pragma unroll
            for (int r = 0; r < 8; ++r) {
                const int row = m0 + wm * 64 + mt * 16 + hi * 8 + r;
                if (row < M) {
                    float v = acc[mt][nt][r];
                    if (bias) v += bias[col];
                    if (act == 1)  // exact GELU
                        v = 0.5f * v * (1.0f + erff(v * 0.70710678118654752f));
                    if (res) v += res[(size_t)row * N + col];
                    if (outF) outF[(size_t)row * N + col] = v;
                    else      outH[(size_t)row * N + col] = (__bf16)v;
                }
            }
        }
    }
}

// ---------------------------------------------------------------------------
// f32 -> bf16 cast
// ---------------------------------------------------------------------------
__global__ void cast_f32_bf16(const float* __restrict__ in, __bf16* __restrict__ out, size_t n)
{
    size_t i = (size_t)blockIdx.x * 256 + threadIdx.x;
    if (i < n) out[i] = (__bf16)in[i];
}

// ---------------------------------------------------------------------------
// Transposing cast: in f32 [R,C] row-major -> out bf16 [C,R] row-major.
// 32x32 LDS tile, 256 threads (32x8). R, C are multiples of 32 here.
// ---------------------------------------------------------------------------
__launch_bounds__(256)
__global__ void cast_transpose_f32_bf16(const float* __restrict__ in,
                                        __bf16* __restrict__ out, int R, int C)
{
    __shared__ float tile[32][33];
    const int r0 = blockIdx.y * 32, c0 = blockIdx.x * 32;
    const int tx = threadIdx.x & 31, ty = threadIdx.x >> 5;
    for (int i = ty; i < 32; i += 8)
        tile[i][tx] = in[(size_t)(r0 + i) * C + (c0 + tx)];
    __syncthreads();
    for (int i = ty; i < 32; i += 8)
        out[(size_t)(c0 + i) * R + (r0 + tx)] = (__bf16)tile[tx][i];
}

// ---------------------------------------------------------------------------
// Per-token RMSNorm over W channels: out_bf16 = in * rsqrt(mean(in^2)+eps) * w
// ---------------------------------------------------------------------------
__launch_bounds__(256)
__global__ void rmsnorm_kernel(const float* __restrict__ in, const float* __restrict__ w,
                               __bf16* __restrict__ out, int W)
{
    const size_t t = blockIdx.x;
    const float* row = in + t * W;
    __shared__ float red[256];
    float ss = 0.f;
    for (int c = threadIdx.x; c < W; c += 256) { float v = row[c]; ss += v * v; }
    red[threadIdx.x] = ss;
    __syncthreads();
    for (int s = 128; s > 0; s >>= 1) {
        if ((int)threadIdx.x < s) red[threadIdx.x] += red[threadIdx.x + s];
        __syncthreads();
    }
    const float r = rsqrtf(red[0] / (float)W + 1e-6f);
    for (int c = threadIdx.x; c < W; c += 256)
        out[t * W + c] = (__bf16)(row[c] * r * w[c]);
}

// ---------------------------------------------------------------------------
// Build padded+rolled branch inputs from xi (B,4096,512):
//   x{1,2}p[b,t',:] = xi[b,(t'+shift)%4100, half] if src<4096 else 0
// ---------------------------------------------------------------------------
__global__ void build_branch(const __bf16* __restrict__ xi,
                             __bf16* __restrict__ x1p, __bf16* __restrict__ x2p)
{
    const size_t total = (size_t)8 * 4100 * 512;
    size_t idx = (size_t)blockIdx.x * 256 + threadIdx.x;
    if (idx >= total) return;
    int c = (int)(idx % 512);
    int t = (int)((idx / 512) % 4100);
    int b = (int)(idx / ((size_t)512 * 4100));
    int shift = (c < 256) ? 5 : 10;
    int src = (t + shift) % 4100;
    __bf16 v = (__bf16)0.0f;
    if (src < 4096) v = xi[((size_t)b * 4096 + src) * 512 + c];
    if (c < 256) x1p[((size_t)b * 4100 + t) * 256 + c]         = v;
    else         x2p[((size_t)b * 4100 + t) * 256 + (c - 256)] = v;
}

// ---------------------------------------------------------------------------
// Windowed attention (one block per (batch, window); 8 heads, hd=32, C=256).
// qkv is fp32 (B,4100,768) [q|k|v], rpb is (2*win-1, 8).
// Mask applies only in the last window (shifted-window wrap regions).
// ---------------------------------------------------------------------------
__launch_bounds__(128)
__global__ void attn_win_kernel(const float* __restrict__ qkv,
                                __bf16* __restrict__ outp,
                                const float* __restrict__ rpb,
                                int win, int shift, int nW)
{
    __shared__ float sq[20 * 256], sk[20 * 256], sv[20 * 256];
    __shared__ float sc[20 * 20];

    const int b = blockIdx.x / nW;
    const int w = blockIdx.x % nW;
    const size_t base = ((size_t)b * 4100 + (size_t)w * win) * 768;

    const int tot = win * 768;
    for (int i = threadIdx.x; i < tot; i += 128) {
        int tok = i / 768, ch = i % 768;
        float v = qkv[base + (size_t)tok * 768 + ch];
        if (ch < 256)      sq[tok * 256 + ch]       = v;
        else if (ch < 512) sk[tok * 256 + ch - 256] = v;
        else               sv[tok * 256 + ch - 512] = v;
    }
    __syncthreads();

    const bool last = (w == nW - 1);
    const float scale = 0.17677669529663687f;   // 32^-0.5
    const int cut = win - shift;

    for (int h = 0; h < 8; ++h) {
        for (int idx = threadIdx.x; idx < win * win; idx += 128) {
            int i = idx / win, j = idx % win;
            const float* qi = sq + i * 256 + h * 32;
            const float* kj = sk + j * 256 + h * 32;
            float s = 0.f;
#pragma unroll
            for (int d = 0; d < 32; ++d) s += qi[d] * kj[d];
            s = s * scale + rpb[(i - j + win - 1) * 8 + h];
            if (last && ((i < cut) != (j < cut))) s -= 100.0f;
            sc[idx] = s;
        }
        __syncthreads();
        for (int i = threadIdx.x; i < win; i += 128) {
            float mx = -1e30f;
            for (int j = 0; j < win; ++j) mx = fmaxf(mx, sc[i * win + j]);
            float sum = 0.f;
            for (int j = 0; j < win; ++j) { float e = __expf(sc[i * win + j] - mx); sc[i * win + j] = e; sum += e; }
            float inv = 1.0f / sum;
            for (int j = 0; j < win; ++j) sc[i * win + j] *= inv;
        }
        __syncthreads();
        for (int idx = threadIdx.x; idx < win * 32; idx += 128) {
            int i = idx / 32, d = idx % 32;
            float o = 0.f;
            for (int j = 0; j < win; ++j) o += sc[i * win + j] * sv[j * 256 + h * 32 + d];
            outp[((size_t)b * 4100 + (size_t)w * win + i) * 256 + h * 32 + d] = (__bf16)o;
        }
        __syncthreads();
    }
}

// ---------------------------------------------------------------------------
// Unroll (+shift) + crop + concat the two projected branches into oc (B,4096,512)
// ---------------------------------------------------------------------------
__global__ void gather_concat(const __bf16* __restrict__ o1,
                              const __bf16* __restrict__ o2,
                              __bf16* __restrict__ oc)
{
    const size_t total = (size_t)8 * 4096 * 512;
    size_t idx = (size_t)blockIdx.x * 256 + threadIdx.x;
    if (idx >= total) return;
    int c = (int)(idx % 512);
    int t = (int)((idx / 512) % 4096);
    int b = (int)(idx / ((size_t)512 * 4096));
    __bf16 v;
    if (c < 256) { int s = (t - 5  + 4100) % 4100; v = o1[((size_t)b * 4100 + s) * 256 + c]; }
    else         { int s = (t - 10 + 4100) % 4100; v = o2[((size_t)b * 4100 + s) * 256 + c - 256]; }
    oc[idx] = v;
}

// ---------------------------------------------------------------------------
// Launcher
// ---------------------------------------------------------------------------
extern "C" void kernel_launch(void* const* d_in, const int* in_sizes, int n_in,
                              void* d_out, int out_size, void* d_ws, size_t ws_size,
                              hipStream_t stream)
{
    (void)in_sizes; (void)n_in; (void)out_size; (void)ws_size;

    const float* x       = (const float*)d_in[0];
    const float* down_w  = (const float*)d_in[1];
    const float* down_b  = (const float*)d_in[2];
    const float* up_w    = (const float*)d_in[3];
    const float* up_b    = (const float*)d_in[4];
    const float* norm1_w = (const float*)d_in[5];
    const float* norm2_w = (const float*)d_in[6];
    const float* qkv1_w  = (const float*)d_in[7];
    const float* qkv1_b  = (const float*)d_in[8];
    const float* proj1_w = (const float*)d_in[9];
    const float* proj1_b = (const float*)d_in[10];
    const float* rpb1    = (const float*)d_in[11];
    const float* qkv2_w  = (const float*)d_in[12];
    const float* qkv2_b  = (const float*)d_in[13];
    const float* proj2_w = (const float*)d_in[14];
    const float* proj2_b = (const float*)d_in[15];
    const float* rpb2    = (const float*)d_in[16];
    const float* ffn_w1  = (const float*)d_in[17];
    const float* ffn_b1  = (const float*)d_in[18];
    const float* ffn_w2  = (const float*)d_in[19];
    const float* ffn_b2  = (const float*)d_in[20];
    float* out           = (float*)d_out;

    char* ws = (char*)d_ws;
    // --- workspace layout (bytes); weights stored TRANSPOSED as bf16 [N,K] ---
    const size_t OFF_WDOWN = 0;                 // 512x1024 bf16
    const size_t OFF_WUP   = 1048576;           // 1024x512 bf16
    const size_t OFF_WQKV1 = 2097152;           // 768x256 bf16
    const size_t OFF_WQKV2 = 2490368;
    const size_t OFF_WPROJ1= 2883584;           // 256x256 bf16
    const size_t OFF_WPROJ2= 3014656;
    const size_t OFF_WFFN1 = 3145728;           // 2048x1024 bf16
    const size_t OFF_WFFN2 = 7340032;
    const size_t OFF_A     = 16777216;          // xb (bf16 x) -> later h (bf16)
    const size_t OFF_Y     = 83886080;          // y f32 (B,T,512) -> later xnew f32 (B,T,1024)
    const size_t OFF_XI    = 218103808;         // xi bf16 (B,T,512) -> later oc bf16
    const size_t OFF_X1P   = 251658240;         // x1p bf16 -> later o1p bf16
    const size_t OFF_X2P   = 268451840;         // x2p bf16 -> later o2p bf16
    const size_t OFF_O1R   = 285245440;         // proj1 out bf16
    const size_t OFF_O2R   = 302039040;         // proj2 out bf16
    const size_t OFF_QKV1  = 318832640;         // qkv1 f32 -> later ffn hidden bf16
    const size_t OFF_QKV2  = 419594240;         // qkv2 f32

    __bf16* wt_down  = (__bf16*)(ws + OFF_WDOWN);
    __bf16* wt_up    = (__bf16*)(ws + OFF_WUP);
    __bf16* wt_qkv1  = (__bf16*)(ws + OFF_WQKV1);
    __bf16* wt_qkv2  = (__bf16*)(ws + OFF_WQKV2);
    __bf16* wt_proj1 = (__bf16*)(ws + OFF_WPROJ1);
    __bf16* wt_proj2 = (__bf16*)(ws + OFF_WPROJ2);
    __bf16* wt_ffn1  = (__bf16*)(ws + OFF_WFFN1);
    __bf16* wt_ffn2  = (__bf16*)(ws + OFF_WFFN2);
    __bf16* xb      = (__bf16*)(ws + OFF_A);
    __bf16* h       = (__bf16*)(ws + OFF_A);      // reuse
    float*  y       = (float* )(ws + OFF_Y);
    float*  xnew    = (float* )(ws + OFF_Y);      // reuse (y dead by then)
    __bf16* xi      = (__bf16*)(ws + OFF_XI);
    __bf16* oc      = (__bf16*)(ws + OFF_XI);     // reuse
    __bf16* x1p     = (__bf16*)(ws + OFF_X1P);
    __bf16* o1p     = (__bf16*)(ws + OFF_X1P);    // reuse
    __bf16* x2p     = (__bf16*)(ws + OFF_X2P);
    __bf16* o2p     = (__bf16*)(ws + OFF_X2P);    // reuse
    __bf16* o1r     = (__bf16*)(ws + OFF_O1R);
    __bf16* o2r     = (__bf16*)(ws + OFF_O2R);
    float*  qkv1    = (float* )(ws + OFF_QKV1);
    float*  qkv2    = (float* )(ws + OFF_QKV2);
    __bf16* hid     = (__bf16*)(ws + OFF_QKV1);   // reuse qkv1+qkv2 region

    const int M  = 8 * 4096;   // 32768 tokens
    const int Mp = 8 * 4100;   // 32800 padded tokens

    auto castT = [&](const float* src, __bf16* dst, int R, int C) {
        cast_transpose_f32_bf16<<<dim3(C / 32, R / 32), 256, 0, stream>>>(src, dst, R, C);
    };

    // --- activation cast + transposed weight casts to bf16 ---
    {
        size_t n = (size_t)M * 1024;
        cast_f32_bf16<<<(unsigned)((n + 255) / 256), 256, 0, stream>>>(x, xb, n);
    }
    castT(down_w,  wt_down,  1024, 512);
    castT(up_w,    wt_up,    512, 1024);
    castT(qkv1_w,  wt_qkv1,  256, 768);
    castT(qkv2_w,  wt_qkv2,  256, 768);
    castT(proj1_w, wt_proj1, 256, 256);
    castT(proj2_w, wt_proj2, 256, 256);
    castT(ffn_w1,  wt_ffn1,  1024, 2048);
    castT(ffn_w2,  wt_ffn2,  2048, 1024);

    // K1: y = x @ down_w + down_b   (f32 out)
    gemm_bf16_kernel<<<dim3(512 / 128, (M + 127) / 128), 128, 0, stream>>>(
        xb, wt_down, down_b, nullptr, y, nullptr, M, 512, 1024, 0);

    // K2: xi = rmsnorm(y, norm1_w)  (bf16)
    rmsnorm_kernel<<<M, 256, 0, stream>>>(y, norm1_w, xi, 512);

    // K3: pad+roll into x1p / x2p
    {
        size_t tot = (size_t)8 * 4100 * 512;
        build_branch<<<(unsigned)((tot + 255) / 256), 256, 0, stream>>>(xi, x1p, x2p);
    }

    // K4/K5: qkv GEMMs (f32 out)
    gemm_bf16_kernel<<<dim3(768 / 128, (Mp + 127) / 128), 128, 0, stream>>>(
        x1p, wt_qkv1, qkv1_b, nullptr, qkv1, nullptr, Mp, 768, 256, 0);
    gemm_bf16_kernel<<<dim3(768 / 128, (Mp + 127) / 128), 128, 0, stream>>>(
        x2p, wt_qkv2, qkv2_b, nullptr, qkv2, nullptr, Mp, 768, 256, 0);

    // K6/K7: windowed attention (bf16 out, overwrites x1p/x2p)
    attn_win_kernel<<<8 * 410, 128, 0, stream>>>(qkv1, o1p, rpb1, 10, 5, 410);
    attn_win_kernel<<<8 * 205, 128, 0, stream>>>(qkv2, o2p, rpb2, 20, 10, 205);

    // K8/K9: projection GEMMs (bf16 out)
    gemm_bf16_kernel<<<dim3(256 / 128, (Mp + 127) / 128), 128, 0, stream>>>(
        o1p, wt_proj1, proj1_b, nullptr, nullptr, o1r, Mp, 256, 256, 0);
    gemm_bf16_kernel<<<dim3(256 / 128, (Mp + 127) / 128), 128, 0, stream>>>(
        o2p, wt_proj2, proj2_b, nullptr, nullptr, o2r, Mp, 256, 256, 0);

    // K10: unroll + crop + concat -> oc (bf16)
    {
        size_t tot = (size_t)M * 512;
        gather_concat<<<(unsigned)((tot + 255) / 256), 256, 0, stream>>>(o1r, o2r, oc);
    }

    // K11: xnew = x + (oc @ up_w + up_b)   (f32 out)
    gemm_bf16_kernel<<<dim3(1024 / 128, (M + 127) / 128), 128, 0, stream>>>(
        oc, wt_up, up_b, x, xnew, nullptr, M, 1024, 512, 0);

    // K12: h = rmsnorm(xnew, norm2_w)  (bf16, reuses xb region)
    rmsnorm_kernel<<<M, 256, 0, stream>>>(xnew, norm2_w, h, 1024);

    // K13: hid = gelu(h @ ffn_w1 + ffn_b1)  (bf16)
    gemm_bf16_kernel<<<dim3(2048 / 128, (M + 127) / 128), 128, 0, stream>>>(
        h, wt_ffn1, ffn_b1, nullptr, nullptr, hid, M, 2048, 1024, 1);

    // K14: out = xnew + (hid @ ffn_w2 + ffn_b2)  (f32 out)
    gemm_bf16_kernel<<<dim3(1024 / 128, (M + 127) / 128), 128, 0, stream>>>(
        hid, wt_ffn2, ffn_b2, xnew, out, nullptr, M, 1024, 2048, 0);
}
